// SubjectLayer_10419590660547
// MI455X (gfx1250) — matrix operand order repaired and checked by reading, not compiled, not run on hardware
//
#include <hip/hip_runtime.h>
#include <stdint.h>

// Problem sizes (fixed by the reference)
#define C_DIM 270
#define T_DIM 2048
#define B_DIM 128

#define K_CHUNK 32             // K per bf16 WMMA
#define M_TILES 17             // ceil(270/16)
#define M_PAD   (M_TILES * 16) // 272
#define N_TILE  128            // columns per workgroup
#define ROW_BF16 40            // LDS row stride in bf16 (32 data + 8 pad) = 80B, 16B aligned

// Wave grid: 2 (M) x 4 (N). waveM=0 -> tiles 0..8, waveM=1 -> tiles 8..16.
// Tile 8 is computed by both halves (identical values, identical columns) so the
// hot loop has NO control flow. 9 tiles x 2 N-subtiles (32 cols) per wave.
#define MT_PER_WAVE 9

typedef __attribute__((ext_vector_type(16))) __bf16 v16bf;
typedef __attribute__((ext_vector_type(8)))  float  v8f;

union alignas(32) FragU {
    uint4  q[2];
    v16bf  v;
};

__device__ inline void split_bf16(float x, __bf16& hi, __bf16& lo) {
    hi = (__bf16)x;                 // RNE to 8-bit mantissa
    lo = (__bf16)(x - (float)hi);   // residual: next ~8 bits
}

__global__ __launch_bounds__(256)
void subject_layer_kernel(const float* __restrict__ x,
                          const int*   __restrict__ sidx,
                          const float* __restrict__ W,
                          const float* __restrict__ bias,
                          float*       __restrict__ out)
{
    __shared__ __bf16 Ahi[M_PAD * ROW_BF16];   // W chunk, high parts  [m][k]
    __shared__ __bf16 Alo[M_PAD * ROW_BF16];   // W chunk, low parts
    __shared__ __bf16 Bhi[N_TILE * ROW_BF16];  // x chunk (transposed) [n][k]
    __shared__ __bf16 Blo[N_TILE * ROW_BF16];

    const int b     = blockIdx.y;
    const int t0    = blockIdx.x * N_TILE;
    const int tid   = threadIdx.x;
    const int lane  = tid & 31;
    const int wave  = tid >> 5;        // 0..7
    const int waveM = wave >> 2;       // 0..1 : which (overlapping) M-tile range
    const int waveN = wave & 3;        // 0..3 : which 32-column slab
    const int l16   = lane & 15;
    const int lh    = (lane >> 4) & 1; // lane half

    const int s = sidx[b];
    const float* Wbase = W + (size_t)s * C_DIM * C_DIM;
    const float* xbase = x + (size_t)b * C_DIM * T_DIM + t0;

    // This wave's M-tiles: mtBase .. mtBase+8 (always 9, ranges overlap at tile 8)
    const int mtBase = waveM * 8;

    v8f acc[MT_PER_WAVE][2];
#pragma unroll
    for (int i = 0; i < MT_PER_WAVE; ++i)
#pragma unroll
        for (int sj = 0; sj < 2; ++sj)
#pragma unroll
            for (int j = 0; j < 8; ++j) acc[i][sj][j] = 0.0f;

    for (int k0 = 0; k0 < C_DIM; k0 += K_CHUNK) {
        __syncthreads();  // previous chunk's LDS reads complete before overwrite

        // ---- stage W[m=0..271][k=0..31] as hi/lo bf16 (coalesced 128B/wave) ----
        {
            const int kk    = tid & 31;
            const int kglob = k0 + kk;
            const bool kval = (kglob < C_DIM);
#pragma unroll
            for (int mb = 0; mb < M_PAD; mb += 8) {
                const int m = mb + (tid >> 5);
                float v = 0.0f;
                if (kval && m < C_DIM) v = Wbase[(size_t)m * C_DIM + kglob];
                __bf16 hi, lo; split_bf16(v, hi, lo);
                Ahi[m * ROW_BF16 + kk] = hi;
                Alo[m * ROW_BF16 + kk] = lo;
            }
        }
        // ---- stage x[k=0..31][n=0..127] transposed into LDS as [n][k] ----
        {
            const int n = tid & 127;
#pragma unroll
            for (int kk = (tid >> 7); kk < K_CHUNK; kk += 2) {
                const int kglob = k0 + kk;
                float v = 0.0f;
                if (kglob < C_DIM) v = xbase[(size_t)kglob * T_DIM + n];
                __bf16 hi, lo; split_bf16(v, hi, lo);
                Bhi[n * ROW_BF16 + kk] = hi;
                Blo[n * ROW_BF16 + kk] = lo;
            }
        }
        __syncthreads();

        // ---- B fragments for this wave's two 16-col subtiles (held all chunk):
        //      column = waveN*32 + sub*16 + l16; lanes 0-15 K={0..7,16..23}, 16-31 K={8..15,24..31}
        FragU fbh[2], fbl[2];
#pragma unroll
        for (int sub = 0; sub < 2; ++sub) {
            const int brow = waveN * 32 + sub * 16 + l16;
            const __bf16* bhp = &Bhi[brow * ROW_BF16 + lh * 8];
            const __bf16* blp = &Blo[brow * ROW_BF16 + lh * 8];
            fbh[sub].q[0] = *(const uint4*)(bhp);
            fbh[sub].q[1] = *(const uint4*)(bhp + 16);
            fbl[sub].q[0] = *(const uint4*)(blp);
            fbl[sub].q[1] = *(const uint4*)(blp + 16);
        }

        // ---- 2-deep software pipeline over A fragments:
        //      issue tile mt+1's 4 ds_load_b128 before tile mt's 6 WMMAs.
        //      A row = tile*16 + l16; K runs {0..7,16..23} / {8..15,24..31}.
        FragU fa[2][2];   // [buffer][hi=0 / lo=1]
        {
            const int arow0 = mtBase * 16 + l16;
            const __bf16* ahp = &Ahi[arow0 * ROW_BF16 + lh * 8];
            const __bf16* alp = &Alo[arow0 * ROW_BF16 + lh * 8];
            fa[0][0].q[0] = *(const uint4*)(ahp);
            fa[0][0].q[1] = *(const uint4*)(ahp + 16);
            fa[0][1].q[0] = *(const uint4*)(alp);
            fa[0][1].q[1] = *(const uint4*)(alp + 16);
        }
#pragma unroll
        for (int mt = 0; mt < MT_PER_WAVE; ++mt) {
            const int cur = mt & 1;
            if (mt + 1 < MT_PER_WAVE) {
                const int arow = (mtBase + mt + 1) * 16 + l16;
                const __bf16* ahp = &Ahi[arow * ROW_BF16 + lh * 8];
                const __bf16* alp = &Alo[arow * ROW_BF16 + lh * 8];
                fa[cur ^ 1][0].q[0] = *(const uint4*)(ahp);
                fa[cur ^ 1][0].q[1] = *(const uint4*)(ahp + 16);
                fa[cur ^ 1][1].q[0] = *(const uint4*)(alp);
                fa[cur ^ 1][1].q[1] = *(const uint4*)(alp + 16);
            }
#pragma unroll
            for (int sub = 0; sub < 2; ++sub) {
                // bf16x3 fp32-emulation: hi*hi + hi*lo + lo*hi
                acc[mt][sub] = __builtin_amdgcn_wmma_f32_16x16x32_bf16(
                    false, fa[cur][0].v, false, fbh[sub].v, (short)0, acc[mt][sub], false, false);
                acc[mt][sub] = __builtin_amdgcn_wmma_f32_16x16x32_bf16(
                    false, fa[cur][0].v, false, fbl[sub].v, (short)0, acc[mt][sub], false, false);
                acc[mt][sub] = __builtin_amdgcn_wmma_f32_16x16x32_bf16(
                    false, fa[cur][1].v, false, fbh[sub].v, (short)0, acc[mt][sub], false, false);
            }
        }
    }

    // ---- epilogue: C/D layout => VGPR r holds M = tile*16 + lh*8 + r, N = l16
    // Duplicate stores for tile 8 (both waveM halves) write identical values.
    const float* bbase = bias + (size_t)s * C_DIM;
    float* outb = out + (size_t)b * C_DIM * T_DIM;
#pragma unroll
    for (int mt = 0; mt < MT_PER_WAVE; ++mt) {
        const int tile = mtBase + mt;
        const int mrow = tile * 16 + lh * 8;
        if (mrow + 7 < C_DIM) {            // uniform-per-lane-half fast path
#pragma unroll
            for (int sub = 0; sub < 2; ++sub) {
                const int ncol = t0 + waveN * 32 + sub * 16 + l16;
                float* op = outb + (size_t)mrow * T_DIM + ncol;
#pragma unroll
                for (int r = 0; r < 8; ++r) {
                    op[(size_t)r * T_DIM] = acc[mt][sub][r] + bbase[mrow + r];
                }
            }
        } else {                            // only tile 16, upper lane-half rows 270..271
#pragma unroll
            for (int sub = 0; sub < 2; ++sub) {
                const int ncol = t0 + waveN * 32 + sub * 16 + l16;
#pragma unroll
                for (int r = 0; r < 8; ++r) {
                    const int m = mrow + r;
                    if (m < C_DIM) {
                        outb[(size_t)m * T_DIM + ncol] = acc[mt][sub][r] + bbase[m];
                    }
                }
            }
        }
    }
}

extern "C" void kernel_launch(void* const* d_in, const int* in_sizes, int n_in,
                              void* d_out, int out_size, void* d_ws, size_t ws_size,
                              hipStream_t stream) {
    const float* x    = (const float*)d_in[0];
    const int*   sidx = (const int*)  d_in[1];
    const float* W    = (const float*)d_in[2];
    const float* bias = (const float*)d_in[3];
    float* out = (float*)d_out;

    dim3 grid(T_DIM / N_TILE, B_DIM);   // 16 column-tiles x 128 samples
    subject_layer_kernel<<<grid, 256, 0, stream>>>(x, sidx, W, bias, out);
}
